// ptcld_GNN_64476049048190
// MI455X (gfx1250) — matrix operand order repaired and checked by reading, not compiled, not run on hardware
//
#include <hip/hip_runtime.h>
#include <hip/hip_bf16.h>

typedef __attribute__((ext_vector_type(4)))  _Float16 v4h;
typedef __attribute__((ext_vector_type(8)))  _Float16 v8h;
typedef __attribute__((ext_vector_type(16))) _Float16 v16h;
typedef __attribute__((ext_vector_type(8)))  float    v8f;

#define BB   32
#define PP   2048
#define KK   20
#define HH   128
#define MID  256
#define NN   (BB * PP)
#define EPSB 1e-5f

// ---------------------------------------------------------------------------
// Pack a row-major f32 weight matrix [Kdim x Ncols] into the wave32 WMMA
// B-fragment layout for v_wmma_f32_16x16x32_f16:
//   chunk(kb, nt): for lane 0..31, 16 consecutive halves
//     b[j] = W[kb*32 + (lane>=16 ? 8:0) + (j<8 ? j : j+8)][nt*16 + lane%16]
// so each lane fetches its whole fragment with one aligned 32B load.
// ---------------------------------------------------------------------------
__global__ __launch_bounds__(256) void pack_b_kernel(const float* __restrict__ src,
                                                     _Float16* __restrict__ dst,
                                                     int Kdim, int Ncols) {
    int p = blockIdx.x * 256 + threadIdx.x;
    if (p >= Kdim * Ncols) return;
    int NT   = Ncols >> 4;
    int j    = p & 15;
    int lane = (p >> 4) & 31;
    int nt   = (p >> 9) % NT;
    int kb   = p / (512 * NT);
    int k    = kb * 32 + (lane >= 16 ? 8 : 0) + (j < 8 ? j : j + 8);
    int c    = nt * 16 + (lane & 15);
    dst[p]   = (_Float16)src[k * Ncols + c];
}

// ---------------------------------------------------------------------------
// Per-cloud kNN. 8 blocks per cloud, each block caches the whole cloud in LDS.
// One thread per query point; top-20 kept in registers via max-replace with
// fully unrolled (static index) scans so nothing touches scratch.
// ---------------------------------------------------------------------------
__global__ __launch_bounds__(256) void knn_kernel(const float* __restrict__ x,
                                                  int* __restrict__ nbr) {
    __shared__ float sx[PP], sy[PP], sz[PP];
    const int cloud = blockIdx.x >> 3;
    const int base  = cloud * PP;
    for (int i = threadIdx.x; i < PP; i += 256) {
        sx[i] = x[(size_t)(base + i) * 3 + 0];
        sy[i] = x[(size_t)(base + i) * 3 + 1];
        sz[i] = x[(size_t)(base + i) * 3 + 2];
    }
    __syncthreads();

    const int   lq = (blockIdx.x & 7) * 256 + threadIdx.x;
    const float qx = sx[lq], qy = sy[lq], qz = sz[lq];

    float d[KK]; int id[KK];
#pragma unroll
    for (int j = 0; j < KK; ++j) { d[j] = 3.4e38f; id[j] = base + lq; }
    float maxd = 3.4e38f; int maxslot = 0;

    for (int c = 0; c < PP; ++c) {
        float dx = sx[c] - qx, dy = sy[c] - qy, dz = sz[c] - qz;
        float dist = dx * dx + dy * dy + dz * dz;
        if (dist < maxd) {
#pragma unroll
            for (int j = 0; j < KK; ++j)
                if (j == maxslot) { d[j] = dist; id[j] = base + c; }
            maxd = -1.0f;
#pragma unroll
            for (int j = 0; j < KK; ++j)
                if (d[j] > maxd) { maxd = d[j]; maxslot = j; }
        }
    }
    const int q = base + lq;
#pragma unroll
    for (int j = 0; j < KK; ++j) nbr[(size_t)q * KK + j] = id[j];
}

// ---------------------------------------------------------------------------
// h0 = f16(x @ Wt + bt).  One thread per (point, 4 features).
// ---------------------------------------------------------------------------
__global__ __launch_bounds__(256) void transfer_kernel(const float* __restrict__ x,
                                                       const float* __restrict__ Wt,
                                                       const float* __restrict__ bt,
                                                       _Float16* __restrict__ h0) {
    int tid = blockIdx.x * 256 + threadIdx.x;   // NN*32 threads
    int q   = tid >> 5;
    int c0  = (tid & 31) * 4;
    float x0 = x[(size_t)q * 3 + 0];
    float x1 = x[(size_t)q * 3 + 1];
    float x2 = x[(size_t)q * 3 + 2];
    v4h out;
#pragma unroll
    for (int j = 0; j < 4; ++j) {
        int c = c0 + j;
        float v = x0 * Wt[c] + x1 * Wt[HH + c] + x2 * Wt[2 * HH + c] + bt[c];
        out[j] = (_Float16)v;
    }
    *(v4h*)(h0 + (size_t)q * HH + c0) = out;
}

// ---------------------------------------------------------------------------
// Fused GIN layer: gather-max (f16, v_pk_max_num_f16) -> GEMM1(+b1,ReLU)
// -> GEMM2(+b2,BN[,ReLU]).
// Block = 128 points, 8 waves; wave w owns rows [16w,16w+16) through ALL
// compute stages, so only one barrier (after the cooperative index stage) is
// needed. LDS: agg 32KB + C1 64KB + nbr 10KB.
// ---------------------------------------------------------------------------
__global__ __launch_bounds__(256) void gin_layer_kernel(
    const _Float16* __restrict__ h_in, const int* __restrict__ nbr,
    const _Float16* __restrict__ W1p,  const float* __restrict__ b1,
    const _Float16* __restrict__ W2p,  const float* __restrict__ b2,
    const float* __restrict__ g,  const float* __restrict__ be,
    const float* __restrict__ rm, const float* __restrict__ rv,
    _Float16* __restrict__ out_h, float* __restrict__ out_f, int relu_out)
{
    extern __shared__ _Float16 smem[];
    _Float16* s_agg = smem;                        // [128][128] f16
    _Float16* s_c1  = smem + 128 * 128;            // [128][256] f16
    int*      s_nbr = (int*)(smem + 128 * 128 + 128 * 256); // [128][20]

    const int  w       = threadIdx.x >> 5;
    const int  lane    = threadIdx.x & 31;
    const int  lrow    = lane & 15;
    const bool hi      = lane >= 16;
    const int  mbase   = w * 16;
    const int  rowbase = blockIdx.x * 128;

    // ---- stage neighbor indices in LDS (coalesced once per block) ----
    for (int t = threadIdx.x; t < 128 * KK; t += 256)
        s_nbr[t] = nbr[(size_t)rowbase * KK + t];
    __syncthreads();

    // ---- gather-max over 20 neighbors + self ----
    // 16B loads: lanes 0-15 handle row 2i, lanes 16-31 handle row 2i+1.
    {
        const int g2 = hi ? 1 : 0;
        const int c0 = lrow * 8;
        for (int i = 0; i < 8; ++i) {
            const int rloc = mbase + i * 2 + g2;
            const int row  = rowbase + rloc;
            v8h acc = *(const v8h*)(h_in + (size_t)row * HH + c0);
#pragma unroll
            for (int k = 0; k < KK; ++k) {
                int nb = s_nbr[rloc * KK + k];
                v8h nv = *(const v8h*)(h_in + (size_t)nb * HH + c0);
                acc = __builtin_elementwise_max(acc, nv);   // v_pk_max_num_f16
            }
            *(v8h*)(s_agg + rloc * HH + c0) = acc;
        }
    }

    // warm L2 for GEMM2 weights while GEMM1 runs
    __builtin_prefetch(W2p + (size_t)lane * 16, 0, 3);

    // ---- GEMM1: agg[16x128] @ W1[128x256] + b1, ReLU -> s_c1 (f16) ----
    const int arow1 = (mbase + lrow) * HH;
    for (int nt = 0; nt < 16; ++nt) {
        v8f acc = {0.f, 0.f, 0.f, 0.f, 0.f, 0.f, 0.f, 0.f};
#pragma unroll
        for (int kb = 0; kb < 4; ++kb) {
            int k0  = kb * 32 + (hi ? 8 : 0);
            v8h alo = *(const v8h*)(s_agg + arow1 + k0);
            v8h ahi = *(const v8h*)(s_agg + arow1 + k0 + 16);
            v16h a  = __builtin_shufflevector(alo, ahi,
                         0,1,2,3,4,5,6,7,8,9,10,11,12,13,14,15);
            v16h bf = *(const v16h*)(W1p + (((kb * 16 + nt) * 32 + lane) << 4));
            acc = __builtin_amdgcn_wmma_f32_16x16x32_f16(
                      false, a, false, bf, (short)0, acc, false, false);
        }
        const int   col  = nt * 16 + lrow;
        const float bias = b1[col];
#pragma unroll
        for (int r = 0; r < 8; ++r) {
            int   row = mbase + r + (hi ? 8 : 0);
            float v   = fmaxf(acc[r] + bias, 0.0f);
            s_c1[row * MID + col] = (_Float16)v;
        }
    }

    // ---- GEMM2: C1[16x256] @ W2[256x128] + b2, BN (+ReLU) -> global ----
    const int arow2 = (mbase + lrow) * MID;
    for (int nt = 0; nt < 8; ++nt) {
        v8f acc = {0.f, 0.f, 0.f, 0.f, 0.f, 0.f, 0.f, 0.f};
#pragma unroll
        for (int kb = 0; kb < 8; ++kb) {
            int k0  = kb * 32 + (hi ? 8 : 0);
            v8h alo = *(const v8h*)(s_c1 + arow2 + k0);
            v8h ahi = *(const v8h*)(s_c1 + arow2 + k0 + 16);
            v16h a  = __builtin_shufflevector(alo, ahi,
                         0,1,2,3,4,5,6,7,8,9,10,11,12,13,14,15);
            v16h bf = *(const v16h*)(W2p + (((kb * 8 + nt) * 32 + lane) << 4));
            acc = __builtin_amdgcn_wmma_f32_16x16x32_f16(
                      false, a, false, bf, (short)0, acc, false, false);
        }
        const int   col   = nt * 16 + lrow;
        const float scale = g[col] * rsqrtf(rv[col] + EPSB);
        const float shift = be[col] - rm[col] * scale;
        const float bias  = b2[col];
#pragma unroll
        for (int r = 0; r < 8; ++r) {
            int   row = mbase + r + (hi ? 8 : 0);
            float v   = (acc[r] + bias) * scale + shift;
            if (relu_out) v = fmaxf(v, 0.0f);
            size_t o = (size_t)(rowbase + row) * HH + col;
            if (out_h) out_h[o] = (_Float16)v;
            if (out_f) out_f[o] = v;
        }
    }
}

// ---------------------------------------------------------------------------
extern "C" void kernel_launch(void* const* d_in, const int* in_sizes, int n_in,
                              void* d_out, int out_size, void* d_ws, size_t ws_size,
                              hipStream_t stream) {
    const float* x    = (const float*)d_in[0];
    // d_in[1] = batch (unused: clouds are equal-size, derived from index)
    const float* Wt   = (const float*)d_in[2];
    const float* bt   = (const float*)d_in[3];
    const float* W1_0 = (const float*)d_in[4];
    const float* b1_0 = (const float*)d_in[5];
    const float* W2_0 = (const float*)d_in[6];
    const float* b2_0 = (const float*)d_in[7];
    const float* g0   = (const float*)d_in[8];
    const float* be0  = (const float*)d_in[9];
    const float* rm0  = (const float*)d_in[10];
    const float* rv0  = (const float*)d_in[11];
    const float* W1_1 = (const float*)d_in[12];
    const float* b1_1 = (const float*)d_in[13];
    const float* W2_1 = (const float*)d_in[14];
    const float* b2_1 = (const float*)d_in[15];
    const float* g1   = (const float*)d_in[16];
    const float* be1  = (const float*)d_in[17];
    const float* rm1  = (const float*)d_in[18];
    const float* rv1  = (const float*)d_in[19];

    char* ws = (char*)d_ws;
    size_t off = 0;
    int*      nbr  = (int*)(ws + off);      off += (size_t)NN * KK * sizeof(int);
    _Float16* h0   = (_Float16*)(ws + off); off += (size_t)NN * HH * sizeof(_Float16);
    _Float16* h1   = (_Float16*)(ws + off); off += (size_t)NN * HH * sizeof(_Float16);
    _Float16* W10p = (_Float16*)(ws + off); off += (size_t)HH * MID * sizeof(_Float16);
    _Float16* W20p = (_Float16*)(ws + off); off += (size_t)MID * HH * sizeof(_Float16);
    _Float16* W11p = (_Float16*)(ws + off); off += (size_t)HH * MID * sizeof(_Float16);
    _Float16* W21p = (_Float16*)(ws + off);

    // weight packing (tiny)
    pack_b_kernel<<<(HH * MID + 255) / 256, 256, 0, stream>>>(W1_0, W10p, HH, MID);
    pack_b_kernel<<<(MID * HH + 255) / 256, 256, 0, stream>>>(W2_0, W20p, MID, HH);
    pack_b_kernel<<<(HH * MID + 255) / 256, 256, 0, stream>>>(W1_1, W11p, HH, MID);
    pack_b_kernel<<<(MID * HH + 255) / 256, 256, 0, stream>>>(W2_1, W21p, MID, HH);

    // kNN (nbr) and transfer MLP (h0)
    knn_kernel<<<NN / 256, 256, 0, stream>>>(x, nbr);
    transfer_kernel<<<(NN * 32) / 256, 256, 0, stream>>>(x, Wt, bt, h0);

    // fused GIN layers
    const size_t lds_bytes =
        (size_t)(128 * 128 + 128 * 256) * sizeof(_Float16) + 128 * KK * sizeof(int);
    gin_layer_kernel<<<NN / 128, 256, lds_bytes, stream>>>(
        h0, nbr, W10p, b1_0, W20p, b2_0, g0, be0, rm0, rv0,
        h1, nullptr, 1);
    gin_layer_kernel<<<NN / 128, 256, lds_bytes, stream>>>(
        h1, nbr, W11p, b1_1, W21p, b2_1, g1, be1, rm1, rv1,
        nullptr, (float*)d_out, 0);
}
